// CustomConv2D_48189533061569
// MI455X (gfx1250) — matrix operand order repaired and checked by reading, not compiled, not run on hardware
//
#include <hip/hip_runtime.h>
#include <cstddef>

typedef __attribute__((ext_vector_type(16))) _Float16 v16h;
typedef __attribute__((ext_vector_type(8)))  float    v8f;

namespace {
constexpr int kB = 64, kC = 3, kH = 384, kW = 384;
constexpr int kN = 576, kK = 16, kO = 1024;
constexpr int kKdim = kC * kK * kK;   // 768
constexpr int kM    = kB * kN;        // 36864

constexpr int TM  = 128;   // M tile per workgroup
constexpr int TO  = 128;   // O tile per workgroup
constexpr int TK  = 32;    // K step (WMMA K)
constexpr int LDA = 40;    // halfs per LDS row: 32 data + 8 pad (80 B stride)
}

__global__ __launch_bounds__(256)
void patch_embed_wmma(const float* __restrict__ x,
                      const int*   __restrict__ centers,
                      const float* __restrict__ w,
                      const float* __restrict__ bias,
                      float*       __restrict__ out)
{
    __shared__ __align__(16) _Float16 As[TM * LDA];  // 10240 B
    __shared__ __align__(16) _Float16 Bs[TO * LDA];  // 10240 B

    const int tid    = threadIdx.x;
    const int m_base = blockIdx.y * TM;
    const int o_base = blockIdx.x * TO;

    // ---- staging assignment: thread t owns tile-row r, k-segment s (16 elems) ----
    const int r = tid >> 1;   // 0..127
    const int s = tid & 1;    // 0: k 0..15, 1: k 16..31 within the K-step

    // A-row global base: patch start for row m
    const int m  = m_base + r;
    const int bi = m / kN;
    const int ni = m - bi * kN;
    const int cy = centers[(bi * kN + ni) * 2 + 0];
    const int cx = centers[(bi * kN + ni) * 2 + 1];
    const float* abase = x + (size_t)bi * (kC * kH * kW)
                           + (size_t)(cy - kK / 2) * kW + (cx - kK / 2);
    // W-row global base (row = output channel o_base + r), 64B aligned
    const float* wrow = w + (size_t)(o_base + r) * kKdim + s * 16;

    // ---- per-wave WMMA coordinates ----
    const int lane  = tid & 31;
    const int wm    = ((tid >> 5) & 1) * 64;   // wave M offset within tile
    const int wo    = (tid >> 6) * 32;         // wave O offset within tile
    const int lhalf = lane >> 4;               // 0: lanes 0-15, 1: lanes 16-31
    const int lrow  = lane & 15;

    const v8f vzero = {0.f, 0.f, 0.f, 0.f, 0.f, 0.f, 0.f, 0.f};
    v8f acc[4][2];
#pragma unroll
    for (int t = 0; t < 4; ++t)
#pragma unroll
        for (int u = 0; u < 2; ++u) acc[t][u] = vzero;

    for (int kk = 0; kk < kKdim / TK; ++kk) {
        // ---- gather stage: A (patch pixels) ----
        const int kg = kk * TK + s * 16;                 // global k of first elem
        const float* ap = abase + (kg >> 8) * (kH * kW)  // channel
                                + ((kg >> 4) & 15) * kW; // row inside patch
        float af[16];
#pragma unroll
        for (int j = 0; j < 16; ++j) af[j] = ap[j];      // 4B-aligned gather

        // ---- stage: W slab (contiguous, 64B aligned) ----
        const float4* wp = (const float4*)(wrow + kk * TK);
        float4 wf[4];
        wf[0] = wp[0]; wf[1] = wp[1]; wf[2] = wp[2]; wf[3] = wp[3];

        __syncthreads();   // all waves done reading LDS from previous step

        union { _Float16 h[16]; int4 q[2]; } pk;
#pragma unroll
        for (int j = 0; j < 16; ++j) pk.h[j] = (_Float16)af[j];
        {
            int4* dst = (int4*)&As[r * LDA + s * 16];
            dst[0] = pk.q[0]; dst[1] = pk.q[1];
        }
        const float* wfl = (const float*)&wf[0];
#pragma unroll
        for (int j = 0; j < 16; ++j) pk.h[j] = (_Float16)wfl[j];
        {
            int4* dst = (int4*)&Bs[r * LDA + s * 16];
            dst[0] = pk.q[0]; dst[1] = pk.q[1];
        }

        __syncthreads();   // LDS tile visible to all waves

        // ---- fragment loads per ISA 16-bit A/B layouts ----
        v16h afr[4], bfr[2];
#pragma unroll
        for (int t = 0; t < 4; ++t) {
            // lane<16: K 0..7 then 16..23 ; lane>=16: K 8..15 then 24..31
            const _Float16* p = &As[(wm + t * 16 + lrow) * LDA + (lhalf << 3)];
            ((int4*)&afr[t])[0] = *(const int4*)p;
            ((int4*)&afr[t])[1] = *(const int4*)(p + 16);
        }
#pragma unroll
        for (int u = 0; u < 2; ++u) {
            // lane<16: K 0..7 then 16..23 of col n ; lane>=16: K 8..15 then 24..31
            const _Float16* p = &Bs[(wo + u * 16 + lrow) * LDA + (lhalf << 3)];
            ((int4*)&bfr[u])[0] = *(const int4*)p;
            ((int4*)&bfr[u])[1] = *(const int4*)(p + 16);
        }

        // ---- 8 WMMAs: 64x32 wave tile ----
#pragma unroll
        for (int t = 0; t < 4; ++t)
#pragma unroll
            for (int u = 0; u < 2; ++u)
                acc[t][u] = __builtin_amdgcn_wmma_f32_16x16x32_f16(
                    false, afr[t], false, bfr[u],
                    (short)0, acc[t][u], false, false);
    }

    // ---- epilogue: bias + store (C/D layout: VGPR e -> M = e + 8*lhalf, N = lrow) ----
    const float bv0 = bias[o_base + wo + lrow];
    const float bv1 = bias[o_base + wo + 16 + lrow];
#pragma unroll
    for (int t = 0; t < 4; ++t) {
#pragma unroll
        for (int u = 0; u < 2; ++u) {
            const int mr = m_base + wm + t * 16 + lhalf * 8;
            const int oc = o_base + wo + u * 16 + lrow;
            float* op = out + (size_t)mr * kO + oc;
            const float bv = u ? bv1 : bv0;
#pragma unroll
            for (int e = 0; e < 8; ++e)
                op[(size_t)e * kO] = acc[t][u][e] + bv;
        }
    }
}

extern "C" void kernel_launch(void* const* d_in, const int* in_sizes, int n_in,
                              void* d_out, int out_size, void* d_ws, size_t ws_size,
                              hipStream_t stream) {
    (void)in_sizes; (void)n_in; (void)d_ws; (void)ws_size; (void)out_size;
    const float* x       = (const float*)d_in[0];
    const int*   centers = (const int*)  d_in[1];
    const float* w       = (const float*)d_in[2];
    const float* bias    = (const float*)d_in[3];
    float*       out     = (float*)d_out;

    dim3 grid(kO / TO, kM / TM);   // (8, 288) = 2304 workgroups
    patch_embed_wmma<<<grid, 256, 0, stream>>>(x, centers, w, bias, out);
}
